// GraphTransformer_52338471469484
// MI455X (gfx1250) — matrix operand order repaired and checked by reading, not compiled, not run on hardware
//
#include <hip/hip_runtime.h>
#include <hip/hip_bf16.h>
#include <math.h>

// ---------------------------------------------------------------------------
// Sizes (fixed by the reference)
// ---------------------------------------------------------------------------
#define NN   4096
#define FEAT 512
#define DD   64
#define KK   100
#define SLD  112          // padded row stride for S (zeros in cols 100..111)
#define MPAD 128          // padded cluster-row count for the S^T @ adj GEMM
#define LL   3
#define HH_  8
#define TT_  101          // tokens = 1 cls + 100 clusters

typedef __attribute__((ext_vector_type(16))) __bf16 v16bf;
typedef __attribute__((ext_vector_type(8)))  float  v8f;

// ---------------------------------------------------------------------------
// wave32 helpers
// ---------------------------------------------------------------------------
__device__ __forceinline__ float wave_sum32(float v) {
    #pragma unroll
    for (int m = 16; m > 0; m >>= 1) v += __shfl_xor(v, m, 32);
    return v;
}
__device__ __forceinline__ float wave_max32(float v) {
    #pragma unroll
    for (int m = 16; m > 0; m >>= 1) v = fmaxf(v, __shfl_xor(v, m, 32));
    return v;
}
__device__ __forceinline__ float block_reduce256(float v, float* red) {
    int tid = threadIdx.x;
    red[tid] = v; __syncthreads();
    #pragma unroll
    for (int s = 128; s > 0; s >>= 1) {
        if (tid < s) red[tid] += red[tid + s];
        __syncthreads();
    }
    float r = red[0]; __syncthreads();
    return r;
}

__device__ __forceinline__ unsigned int pack_bf16x2(float lo, float hi) {
    union { __bf16 b[2]; unsigned int u; } p;
    p.b[0] = (__bf16)lo;
    p.b[1] = (__bf16)hi;
    return p.u;
}

// ---------------------------------------------------------------------------
// Tiled bf16 WMMA GEMM:  D = A*B (+Cadd), fp32 in/out, bf16 compute.
//
// Block tile 128(M) x 64(N), K-step 32, 256 threads = 8 waves.
// Wave w owns the 16-row strip m = w*16..w*16+15 and all 64 columns:
// one A fragment feeds 4 v_wmma_f32_16x16x32_bf16 per K-step.
//
// LDS is in *fragment order*: each lane's 16 bf16 fragment elements are 32
// contiguous bytes -> fragment build is one v16bf load (2x ds_load_b128).
//
// Staging is register-pipelined AND LDS is ping-pong double-buffered:
//   iter i:  frag-read buf[cur] | reg-store -> buf[nxt] | global-load kk+2*TK
//            4x WMMA | single __syncthreads()
// so there is only ONE barrier per K-step and global latency hides under
// the matrix ops.
//
// A fragment (16x32, ISA layout): lane = half*16+r holds row r,
//   elems e=0..7  -> K = half*8 + e;  e=8..15 -> K = half*8 + 16 + (e-8)
// B fragment (32x16): lane = half*16+r holds col r, elem e -> K ordered as for A.
//
// Requires: M % 128 == 0, N % 64 == 0, K % 64 == 0, lda % 4 == 0.
// TRANS: logical A[m][k] = A[k*lda + m], rows m >= Avalid read as 0.
// ---------------------------------------------------------------------------
#define TM 128
#define TN 64
#define TK 32
#define ASZ (8 * 32 * 16)   // elements per A buffer (8 KB)
#define BSZ (4 * 32 * 16)   // elements per B buffer (4 KB)

template <int TRANS, int ADDC>
__global__ __launch_bounds__(256)
void gemm_bf16_wmma_t(const float* __restrict__ A, int lda, int Avalid,
                      const float* __restrict__ B, int ldb,
                      const float* __restrict__ Cadd, int ldc,
                      float* __restrict__ D, int ldd,
                      int M, int N, int K)
{
    // ping-pong fragment-ordered staging buffers (24 KB total)
    __shared__ __align__(32) __bf16 As[2 * ASZ];
    __shared__ __align__(32) __bf16 Bs[2 * BSZ];

    const int tid  = threadIdx.x;
    const int lane = tid & 31;
    const int wid  = tid >> 5;           // wave id = M-subtile (0..7)
    const int m0   = blockIdx.y * TM;
    const int n0   = blockIdx.x * TN;

    // ---- per-thread staging geometry (constant across K-steps) ----
    // A: 1024 K-quads; qidx = tid + it*256, j = (qidx&7)*4, i = qidx>>3
    int aOffLDS[4]; int aRow[4]; int aCol[4];
    #pragma unroll
    for (int it = 0; it < 4; ++it) {
        int qidx = tid + it * 256;
        int j = (qidx & 7) * 4;
        int i = qidx >> 3;
        int sub = i >> 4, rr = i & 15;
        int hf = (j >> 3) & 1, run = j >> 4, pos = j & 7;
        aOffLDS[it] = sub * 512 + (hf * 16 + rr) * 16 + run * 8 + pos;
        aRow[it] = m0 + i;
        aCol[it] = j;
    }
    // B: 512 K-quads; qidx = tid + it*256, n = qidx&63, kq = (qidx>>6)*4
    int bOffLDS[2]; int bColN[2]; int bK[2];
    #pragma unroll
    for (int it = 0; it < 2; ++it) {
        int qidx = tid + it * 256;
        int n = qidx & 63;
        int kq = (qidx >> 6) * 4;
        int hf = (kq >> 3) & 1, e = (kq >> 4) * 8 + (kq & 7);
        int grp = n >> 4, rr = n & 15;
        bOffLDS[it] = grp * 512 + (hf * 16 + rr) * 16 + e;
        bColN[it] = n0 + n;
        bK[it] = kq;
    }

    uint2 aReg[4];
    uint2 bReg[2];

    auto load_tiles = [&](int kk) {
        #pragma unroll
        for (int it = 0; it < 4; ++it) {
            float v0, v1, v2, v3;
            if (TRANS) {
                int kg = kk + aCol[it];
                int m = aRow[it];
                bool ok = m < Avalid;
                v0 = ok ? A[(size_t)(kg + 0) * lda + m] : 0.f;
                v1 = ok ? A[(size_t)(kg + 1) * lda + m] : 0.f;
                v2 = ok ? A[(size_t)(kg + 2) * lda + m] : 0.f;
                v3 = ok ? A[(size_t)(kg + 3) * lda + m] : 0.f;
            } else {
                const float4 q = *(const float4*)&A[(size_t)aRow[it] * lda + kk + aCol[it]];
                v0 = q.x; v1 = q.y; v2 = q.z; v3 = q.w;
            }
            aReg[it].x = pack_bf16x2(v0, v1);
            aReg[it].y = pack_bf16x2(v2, v3);
        }
        #pragma unroll
        for (int it = 0; it < 2; ++it) {
            const float* bp = &B[(size_t)(kk + bK[it]) * ldb + bColN[it]];
            float v0 = bp[0];
            float v1 = bp[(size_t)ldb];
            float v2 = bp[(size_t)2 * ldb];
            float v3 = bp[(size_t)3 * ldb];
            bReg[it].x = pack_bf16x2(v0, v1);
            bReg[it].y = pack_bf16x2(v2, v3);
        }
    };
    auto store_tiles = [&](int buf) {
        __bf16* ab = As + buf * ASZ;
        __bf16* bb = Bs + buf * BSZ;
        #pragma unroll
        for (int it = 0; it < 4; ++it)
            *(uint2*)&ab[aOffLDS[it]] = aReg[it];
        #pragma unroll
        for (int it = 0; it < 2; ++it)
            *(uint2*)&bb[bOffLDS[it]] = bReg[it];
    };

    v8f c0 = {0.f,0.f,0.f,0.f,0.f,0.f,0.f,0.f};
    v8f c1 = c0, c2 = c0, c3 = c0;

    const v16bf* Afrag = (const v16bf*)As;   // index: buf*256 + subtile*32 + lane
    const v16bf* Bfrag = (const v16bf*)Bs;   // index: buf*128 + ngroup*32 + lane

    // prologue: fill buffer 0, prefetch tile 1 into registers
    load_tiles(0);
    store_tiles(0);
    __syncthreads();
    if (K > TK) load_tiles(TK);

    for (int kk = 0; kk < K; kk += TK) {
        const int cur = (kk >> 5) & 1;

        v16bf a  = Afrag[cur * 256 + wid * 32 + lane];
        v16bf b0 = Bfrag[cur * 128 + 0 * 32 + lane];
        v16bf b1 = Bfrag[cur * 128 + 1 * 32 + lane];
        v16bf b2 = Bfrag[cur * 128 + 2 * 32 + lane];
        v16bf b3 = Bfrag[cur * 128 + 3 * 32 + lane];

        if (kk + TK < K)       store_tiles(cur ^ 1);       // regs -> other buffer
        if (kk + 2 * TK < K) { load_tiles(kk + 2 * TK);    // next-next tile -> regs
            __builtin_prefetch(&B[(size_t)(kk + 2 * TK) * ldb + n0 + lane], 0, 1);
        }

        c0 = __builtin_amdgcn_wmma_f32_16x16x32_bf16(false, a, false, b0,
                                                     (short)0, c0, false, false);
        c1 = __builtin_amdgcn_wmma_f32_16x16x32_bf16(false, a, false, b1,
                                                     (short)0, c1, false, false);
        c2 = __builtin_amdgcn_wmma_f32_16x16x32_bf16(false, a, false, b2,
                                                     (short)0, c2, false, false);
        c3 = __builtin_amdgcn_wmma_f32_16x16x32_bf16(false, a, false, b3,
                                                     (short)0, c3, false, false);

        __syncthreads();   // single barrier per K-step (ping-pong)
    }

    // epilogue: D element i of accumulator g -> row half*8+i, col g*16+r
    const int half = lane >> 4;
    const int r    = lane & 15;
    #pragma unroll
    for (int g = 0; g < 4; ++g) {
        v8f c = (g == 0) ? c0 : (g == 1) ? c1 : (g == 2) ? c2 : c3;
        int nn = n0 + g * 16 + r;
        #pragma unroll
        for (int i = 0; i < 8; ++i) {
            int mm = m0 + wid * 16 + half * 8 + i;
            float v = c[i];
            if (ADDC) v += Cadd[(size_t)mm * ldc + nn];
            D[(size_t)mm * ldd + nn] = v;
        }
    }
}

// ---------------------------------------------------------------------------
// Small kernels
// ---------------------------------------------------------------------------
__global__ void zero_scalars_k(float* scal, int n) {
    int t = threadIdx.x;
    if (t < n) scal[t] = 0.f;
}

// dflat[n] = sum_j adj[n][j]   (one wave per row)
__global__ __launch_bounds__(256)
void dflat_k(const float* __restrict__ adj, float* __restrict__ dflat) {
    int lane = threadIdx.x & 31, wid = threadIdx.x >> 5;
    int row = blockIdx.x * 8 + wid;
    float acc = 0.f;
    for (int j = lane; j < NN; j += 32) acc += adj[(size_t)row * NN + j];
    acc = wave_sum32(acc);
    if (lane == 0) dflat[row] = acc;
}

// per-channel batch stats over Z[4096][64] -> stats[c]=mean, stats[64+c]=invstd
__global__ __launch_bounds__(256)
void bnstats_k(const float* __restrict__ Z, float* __restrict__ stats) {
    __shared__ float red[256];
    int c = blockIdx.x;
    float s = 0.f, sq = 0.f;
    for (int rI = threadIdx.x; rI < NN; rI += 256) {
        float v = Z[(size_t)rI * DD + c];
        s += v; sq += v * v;
    }
    float ts  = block_reduce256(s,  red);
    float tsq = block_reduce256(sq, red);
    if (threadIdx.x == 0) {
        float mean = ts / (float)NN;
        float var  = tsq / (float)NN - mean * mean;
        stats[c]      = mean;
        stats[DD + c] = rsqrtf(var + 1e-5f);
    }
}

// BatchNorm affine + row L2-normalize : Yn = BN(Z) / max(||.||,1e-12)
__global__ __launch_bounds__(256)
void bnl2_k(const float* __restrict__ Z, const float* __restrict__ stats,
            const float* __restrict__ g, const float* __restrict__ b,
            float* __restrict__ Yn) {
    int lane = threadIdx.x & 31, wid = threadIdx.x >> 5;
    int n = blockIdx.x * 8 + wid;
    int c0 = lane, c1 = lane + 32;
    float z0 = (Z[(size_t)n * DD + c0] - stats[c0]) * stats[DD + c0] * g[c0] + b[c0];
    float z1 = (Z[(size_t)n * DD + c1] - stats[c1]) * stats[DD + c1] * g[c1] + b[c1];
    float ss = wave_sum32(z0 * z0 + z1 * z1);
    float inv = 1.f / fmaxf(sqrtf(ss), 1e-12f);
    Yn[(size_t)n * DD + c0] = z0 * inv;
    Yn[(size_t)n * DD + c1] = z1 * inv;
}

// S = softmax(Yn @ pool_w + pool_b) rows; also accumulates
// mincut_den += dflat[n] * sum_k S[n,k]^2 . One wave per row; S padded to SLD.
__global__ __launch_bounds__(256)
void softmaxS_k(const float* __restrict__ Yn, const float* __restrict__ pw,
                const float* __restrict__ pb, const float* __restrict__ dflat,
                float* __restrict__ S, float* __restrict__ scal) {
    int lane = threadIdx.x & 31, wid = threadIdx.x >> 5;
    int n = blockIdx.x * 8 + wid;
    float vals[4];
    #pragma unroll
    for (int slot = 0; slot < 4; ++slot) {
        int k = lane + 32 * slot;
        float acc = -1e30f;
        if (k < KK) {
            acc = pb[k];
            for (int c = 0; c < DD; ++c)
                acc += Yn[(size_t)n * DD + c] * pw[c * KK + k];
        }
        vals[slot] = acc;
    }
    float mx = fmaxf(fmaxf(vals[0], vals[1]), fmaxf(vals[2], vals[3]));
    mx = wave_max32(mx);
    float es[4]; float sum = 0.f;
    #pragma unroll
    for (int slot = 0; slot < 4; ++slot) {
        int k = lane + 32 * slot;
        es[slot] = (k < KK) ? expf(vals[slot] - mx) : 0.f;
        sum += es[slot];
    }
    sum = wave_sum32(sum);
    float inv = 1.f / sum;
    float ssq = 0.f;
    #pragma unroll
    for (int slot = 0; slot < 4; ++slot) {
        int k = lane + 32 * slot;
        if (k < SLD) {
            float s = (k < KK) ? es[slot] * inv : 0.f;   // zero padding cols
            S[(size_t)n * SLD + k] = s;
            ssq += s * s;
        }
    }
    ssq = wave_sum32(ssq);
    if (lane == 0) atomicAdd(&scal[1], dflat[n] * ssq);
}

// mincut_num += sum_n SA[k][n] * S[n][k]   (one block per cluster k)
__global__ __launch_bounds__(256)
void trace_k(const float* __restrict__ SA, const float* __restrict__ S,
             float* __restrict__ scal) {
    __shared__ float red[256];
    int k = blockIdx.x;
    float acc = 0.f;
    for (int n = threadIdx.x; n < NN; n += 256)
        acc += SA[(size_t)k * NN + n] * S[(size_t)n * SLD + k];
    float t = block_reduce256(acc, red);
    if (threadIdx.x == 0) atomicAdd(&scal[0], t);
}

// out[k][d] = sum_n S[n][k] * Yn[n][d]   (one wave per output element)
__global__ __launch_bounds__(256)
void outc_k(const float* __restrict__ S, const float* __restrict__ Yn,
            float* __restrict__ OUTC) {
    int lane = threadIdx.x & 31, wid = threadIdx.x >> 5;
    int id = blockIdx.x * 8 + wid;          // < 100*64
    int k = id >> 6, d = id & 63;
    float acc = 0.f;
    for (int n = lane; n < NN; n += 32)
        acc += S[(size_t)n * SLD + k] * Yn[(size_t)n * DD + d];
    acc = wave_sum32(acc);
    if (lane == 0) OUTC[k * DD + d] = acc;
}

// ss[k][l] = sum_n S[n][k] * S[n][l]
__global__ __launch_bounds__(256)
void ss_k(const float* __restrict__ S, float* __restrict__ ssm) {
    int lane = threadIdx.x & 31, wid = threadIdx.x >> 5;
    int id = blockIdx.x * 8 + wid;
    if (id >= KK * KK) return;
    int k = id / KK, l = id % KK;
    float acc = 0.f;
    for (int n = lane; n < NN; n += 32)
        acc += S[(size_t)n * SLD + k] * S[(size_t)n * SLD + l];
    acc = wave_sum32(acc);
    if (lane == 0) ssm[id] = acc;
}

// ortho_loss = || ss/||ss||_F - I/sqrt(K) ||_F ; mincut_loss = -num/den
__global__ __launch_bounds__(256)
void losses_k(const float* __restrict__ ssm, const float* __restrict__ scal,
              float* __restrict__ out) {
    __shared__ float red[256];
    __shared__ float ssF;
    float acc = 0.f;
    for (int i = threadIdx.x; i < KK * KK; i += 256) acc += ssm[i] * ssm[i];
    float tot = block_reduce256(acc, red);
    if (threadIdx.x == 0) ssF = sqrtf(tot);
    __syncthreads();
    float a2 = 0.f;
    for (int i = threadIdx.x; i < KK * KK; i += 256) {
        int k = i / KK, l = i % KK;
        float d = ssm[i] / ssF - ((k == l) ? 0.1f : 0.f);   // I/||I|| = I/10
        a2 += d * d;
    }
    float tot2 = block_reduce256(a2, red);
    if (threadIdx.x == 0) {
        out[3] = sqrtf(tot2);
        out[2] = -(scal[0] / scal[1]);
    }
}

// ---------------------------------------------------------------------------
// 3-layer transformer over 101 tokens of dim 64 (tiny; single block)
// ---------------------------------------------------------------------------
__global__ __launch_bounds__(256)
void transformer_k(const float* __restrict__ OUTC, const float* __restrict__ cls,
                   const float* __restrict__ ln1g, const float* __restrict__ ln1b,
                   const float* __restrict__ qkvw, const float* __restrict__ qkvb,
                   const float* __restrict__ projw, const float* __restrict__ projb,
                   const float* __restrict__ ln2g, const float* __restrict__ ln2b,
                   const float* __restrict__ fc1w, const float* __restrict__ fc1b,
                   const float* __restrict__ fc2w, const float* __restrict__ fc2b,
                   const float* __restrict__ normg, const float* __restrict__ normb,
                   const float* __restrict__ headw, const float* __restrict__ headb,
                   float* TTb, float* HHb, float* QKV, float* ATT,
                   float* OO, float* FF, float* out)
{
    const int tid = threadIdx.x;
    const float scale = 0.35355339059327378f;   // 1/sqrt(8)

    for (int idx = tid; idx < TT_ * DD; idx += 256) {
        int ti = idx >> 6, d = idx & 63;
        TTb[idx] = (ti == 0) ? cls[d] : OUTC[(ti - 1) * DD + d];
    }
    __syncthreads();

    for (int l = 0; l < LL; ++l) {
        // LN1 -> HHb
        for (int row = tid; row < TT_; row += 256) {
            const float* x = TTb + row * DD;
            float m = 0.f; for (int c = 0; c < DD; ++c) m += x[c]; m /= (float)DD;
            float v = 0.f; for (int c = 0; c < DD; ++c) { float d = x[c] - m; v += d * d; }
            v /= (float)DD;
            float is = rsqrtf(v + 1e-6f);
            for (int c = 0; c < DD; ++c)
                HHb[row * DD + c] = (x[c] - m) * is * ln1g[l * DD + c] + ln1b[l * DD + c];
        }
        __syncthreads();
        // QKV = HHb @ qkv_w + b   [101 x 192]
        for (int idx = tid; idx < TT_ * 192; idx += 256) {
            int ti = idx / 192, j = idx % 192;
            float acc = qkvb[l * 192 + j];
            for (int c = 0; c < DD; ++c)
                acc += HHb[ti * DD + c] * qkvw[((size_t)l * DD + c) * 192 + j];
            QKV[idx] = acc;
        }
        __syncthreads();
        // attention logits  [8][101][101]
        for (int idx = tid; idx < HH_ * TT_ * TT_; idx += 256) {
            int h = idx / (TT_ * TT_), rem = idx % (TT_ * TT_);
            int i = rem / TT_, j = rem % TT_;
            float acc = 0.f;
            for (int d0 = 0; d0 < 8; ++d0)
                acc += QKV[i * 192 + h * 8 + d0] * QKV[j * 192 + 64 + h * 8 + d0];
            ATT[idx] = acc * scale;
        }
        __syncthreads();
        // softmax rows
        for (int row = tid; row < HH_ * TT_; row += 256) {
            float* a = ATT + row * TT_;
            float mx = -1e30f;
            for (int j = 0; j < TT_; ++j) mx = fmaxf(mx, a[j]);
            float s = 0.f;
            for (int j = 0; j < TT_; ++j) { float e = expf(a[j] - mx); a[j] = e; s += e; }
            float inv = 1.f / s;
            for (int j = 0; j < TT_; ++j) a[j] *= inv;
        }
        __syncthreads();
        // O = att @ V
        for (int idx = tid; idx < TT_ * DD; idx += 256) {
            int ti = idx >> 6, c = idx & 63;
            int h = c >> 3, d0 = c & 7;
            float acc = 0.f;
            for (int j = 0; j < TT_; ++j)
                acc += ATT[(h * TT_ + ti) * TT_ + j] * QKV[j * 192 + 128 + h * 8 + d0];
            OO[idx] = acc;
        }
        __syncthreads();
        // t += O @ proj_w + b
        for (int idx = tid; idx < TT_ * DD; idx += 256) {
            int ti = idx >> 6, d = idx & 63;
            float acc = projb[l * DD + d];
            for (int c = 0; c < DD; ++c)
                acc += OO[ti * DD + c] * projw[((size_t)l * DD + c) * DD + d];
            TTb[idx] += acc;
        }
        __syncthreads();
        // LN2 -> HHb
        for (int row = tid; row < TT_; row += 256) {
            const float* x = TTb + row * DD;
            float m = 0.f; for (int c = 0; c < DD; ++c) m += x[c]; m /= (float)DD;
            float v = 0.f; for (int c = 0; c < DD; ++c) { float d = x[c] - m; v += d * d; }
            v /= (float)DD;
            float is = rsqrtf(v + 1e-6f);
            for (int c = 0; c < DD; ++c)
                HHb[row * DD + c] = (x[c] - m) * is * ln2g[l * DD + c] + ln2b[l * DD + c];
        }
        __syncthreads();
        // FF = gelu(HHb @ fc1 + b)  exact gelu
        for (int idx = tid; idx < TT_ * 256; idx += 256) {
            int ti = idx >> 8, j = idx & 255;
            float acc = fc1b[l * 256 + j];
            for (int c = 0; c < DD; ++c)
                acc += HHb[ti * DD + c] * fc1w[((size_t)l * DD + c) * 256 + j];
            FF[idx] = 0.5f * acc * (1.f + erff(acc * 0.70710678118654752f));
        }
        __syncthreads();
        // t += FF @ fc2 + b
        for (int idx = tid; idx < TT_ * DD; idx += 256) {
            int ti = idx >> 6, d = idx & 63;
            float acc = fc2b[l * DD + d];
            for (int c = 0; c < 256; ++c)
                acc += FF[ti * 256 + c] * fc2w[((size_t)l * 256 + c) * DD + d];
            TTb[idx] += acc;
        }
        __syncthreads();
    }

    // final LN (row 0 only feeds the head)
    if (tid == 0) {
        const float* x = TTb;
        float m = 0.f; for (int c = 0; c < DD; ++c) m += x[c]; m /= (float)DD;
        float v = 0.f; for (int c = 0; c < DD; ++c) { float d = x[c] - m; v += d * d; }
        v /= (float)DD;
        float is = rsqrtf(v + 1e-6f);
        for (int c = 0; c < DD; ++c)
            HHb[c] = (x[c] - m) * is * normg[c] + normb[c];
    }
    __syncthreads();
    for (int c = tid; c < 2; c += 256) {
        float acc = headb[c];
        for (int d = 0; d < DD; ++d) acc += HHb[d] * headw[d * 2 + c];
        out[c] = acc;
    }
}

// ---------------------------------------------------------------------------
// Host launcher
// ---------------------------------------------------------------------------
extern "C" void kernel_launch(void* const* d_in, const int* in_sizes, int n_in,
                              void* d_out, int out_size, void* d_ws, size_t ws_size,
                              hipStream_t stream) {
    const float* node_feat = (const float*)d_in[0];
    const float* adj       = (const float*)d_in[1];
    const float* conv_w    = (const float*)d_in[2];
    // d_in[3] conv_b: exactly cancels under BatchNorm mean-subtraction
    const float* bn_g   = (const float*)d_in[4];
    const float* bn_b   = (const float*)d_in[5];
    const float* pool_w = (const float*)d_in[6];
    const float* pool_b = (const float*)d_in[7];
    const float* cls    = (const float*)d_in[8];
    const float* ln1_g  = (const float*)d_in[9];
    const float* ln1_b  = (const float*)d_in[10];
    const float* qkv_w  = (const float*)d_in[11];
    const float* qkv_b  = (const float*)d_in[12];
    const float* proj_w = (const float*)d_in[13];
    const float* proj_b = (const float*)d_in[14];
    const float* ln2_g  = (const float*)d_in[15];
    const float* ln2_b  = (const float*)d_in[16];
    const float* fc1_w  = (const float*)d_in[17];
    const float* fc1_b  = (const float*)d_in[18];
    const float* fc2_w  = (const float*)d_in[19];
    const float* fc2_b  = (const float*)d_in[20];
    const float* norm_g = (const float*)d_in[21];
    const float* norm_b = (const float*)d_in[22];
    const float* head_w = (const float*)d_in[23];
    const float* head_b = (const float*)d_in[24];
    float* out = (float*)d_out;

    char* ws = (char*)d_ws;
    size_t off = 0;
    auto alloc = [&](size_t nElem) -> float* {
        float* p = (float*)(ws + off);
        off += (nElem * sizeof(float) + 255) & ~(size_t)255;
        return p;
    };
    float* Y     = alloc((size_t)NN * FEAT);   // adj@x + x        [4096,512]
    float* Z     = alloc((size_t)NN * DD);     // Y @ conv_w       [4096,64]
    float* Yn    = alloc((size_t)NN * DD);     // BN + L2-norm     [4096,64]
    float* S     = alloc((size_t)NN * SLD);    // softmax assign   [4096,112]
    float* SA    = alloc((size_t)MPAD * NN);   // S^T @ adj        [128,4096]
    float* dflat = alloc(NN);
    float* stats = alloc(2 * DD);
    float* ssm   = alloc(KK * KK);
    float* scal  = alloc(8);                   // [0]=mincut_num [1]=mincut_den
    float* OUTC  = alloc(KK * DD);             // cluster tokens  [100,64]
    float* TTb   = alloc(TT_ * DD);
    float* HHb   = alloc(TT_ * DD);
    float* QKV   = alloc(TT_ * 192);
    float* ATT   = alloc((size_t)HH_ * TT_ * TT_);
    float* OO    = alloc(TT_ * DD);
    float* FF    = alloc(TT_ * 256);
    (void)ws_size; (void)in_sizes; (void)n_in; (void)out_size;

    zero_scalars_k<<<1, 32, 0, stream>>>(scal, 8);
    dflat_k<<<NN / 8, 256, 0, stream>>>(adj, dflat);

    // Y = adj @ x + x   (M=4096, N=512, K=4096)
    gemm_bf16_wmma_t<0, 1><<<dim3(FEAT / TN, NN / TM), 256, 0, stream>>>(
        adj, NN, NN, node_feat, FEAT, node_feat, FEAT,
        Y, FEAT, NN, FEAT, NN);

    // Z = Y @ conv_w    (M=4096, N=64, K=512)
    gemm_bf16_wmma_t<0, 0><<<dim3(DD / TN, NN / TM), 256, 0, stream>>>(
        Y, FEAT, NN, conv_w, DD, nullptr, 0,
        Z, DD, NN, DD, FEAT);

    bnstats_k<<<DD, 256, 0, stream>>>(Z, stats);
    bnl2_k<<<NN / 8, 256, 0, stream>>>(Z, stats, bn_g, bn_b, Yn);
    softmaxS_k<<<NN / 8, 256, 0, stream>>>(Yn, pool_w, pool_b, dflat, S, scal);

    // SA = S^T @ adj    (M=128 padded, N=4096, K=4096), A read transposed
    gemm_bf16_wmma_t<1, 0><<<dim3(NN / TN, MPAD / TM), 256, 0, stream>>>(
        S, SLD, SLD, adj, NN, nullptr, 0,
        SA, NN, MPAD, NN, NN);

    trace_k<<<KK, 256, 0, stream>>>(SA, S, scal);
    outc_k<<<(KK * DD) / 8, 256, 0, stream>>>(S, Yn, OUTC);
    ss_k<<<(KK * KK + 7) / 8, 256, 0, stream>>>(S, ssm);
    losses_k<<<1, 256, 0, stream>>>(ssm, scal, out);

    transformer_k<<<1, 256, 0, stream>>>(
        OUTC, cls, ln1_g, ln1_b, qkv_w, qkv_b, proj_w, proj_b,
        ln2_g, ln2_b, fc1_w, fc1_b, fc2_w, fc2_b, norm_g, norm_b,
        head_w, head_b, TTb, HHb, QKV, ATT, OO, FF, out);
}
